// CAML_75943611728727
// MI455X (gfx1250) — compile-verified, matrix operand order checked by reading, not run on hardware
//
#include <hip/hip_runtime.h>
#include <hip/hip_bf16.h>

// ---------------- problem constants ----------------
#define VOCAB 50000
#define E     256
#define KK    512      // num_kernels (channels out of conv)
#define KS    9
#define LL    8192     // labels
#define BB    8
#define TT    2048
#define PAD   4
#define TPAD  (TT + KS - 1)          // 2056
#define JTOT  (E * KS)               // 2304 im2col depth
#define JCH   (JTOT / 32)            // 72 bf16 wmma chunks
#define KCH   (KK / 32)              // 16 bf16 wmma chunks over channel dim

// LDS tile layout for the attention kernel: 16 rows x 512 bf16, with the TDM
// pad feature inserting 16B after each 1024B row -> 1040B row stride.
// Bank math: row stride = 260 dwords; 260 % 64 = 4, so the 16 B-columns map
// to banks {0,4,8,...,60}, and each ds_load_b128 covers 4 consecutive banks:
// all 64 banks busy, zero conflicts.
#define LROW   520                    // row stride in bf16 elements (1040 B)
#define LTILE  (16 * LROW)            // 8320 elements = 16640 B per buffer

typedef __attribute__((ext_vector_type(16))) __bf16 v16bf;
typedef __attribute__((ext_vector_type(8)))  float  v8f;
typedef __attribute__((ext_vector_type(4)))  unsigned int u32x4;
typedef __attribute__((ext_vector_type(8)))  int i32x8;
typedef __attribute__((ext_vector_type(4)))  int i32x4;

union FragU { v16bf v; uint4 q[2]; };

// float -> bf16 raw bits, round-to-nearest-even
__device__ __forceinline__ unsigned short f2bf(float x) {
  unsigned int u = __float_as_uint(x);
  unsigned int r = u + 0x7FFFu + ((u >> 16) & 1u);
  return (unsigned short)(r >> 16);
}

// Load one 16x32 bf16 WMMA operand fragment for this lane.
//   halves 0..7  hold K = kbase + hi*8 + {0..7}
//   halves 8..15 hold K = kbase + 16 + hi*8 + {0..7}
__device__ __forceinline__ v16bf load_frag(const unsigned short* __restrict__ p,
                                           int kbase, int hi) {
  FragU f;
  f.q[0] = *reinterpret_cast<const uint4*>(p + kbase + hi * 8);
  f.q[1] = *reinterpret_cast<const uint4*>(p + kbase + 16 + hi * 8);
  return f.v;
}

// Issue a TDM DMA of one feats tile [16 rows x 512 bf16] into LDS at ldsoff.
// D# per CDNA5 ISA ch.8: group0 = {count|flags, lds_addr, global_addr, type},
// group1 = {mask/data_size/pad, dims, tile dims, strides}. 2D tensor, so the
// remaining descriptor groups are zero. Tracked with TENSORcnt.
// This toolchain exposes the 6-arg builtin: (g0, g1, g2, g3, g4, cpol).
__device__ __forceinline__ void tdm_load_tile(const unsigned short* gsrc,
                                              unsigned ldsoff) {
  unsigned long long ga = (unsigned long long)(uintptr_t)gsrc;
  u32x4 g0;
  g0[0] = 1u;                                       // count=1, user mode
  g0[1] = ldsoff;                                   // lds_addr (bytes)
  g0[2] = (unsigned)ga;                             // global_addr[31:0]
  g0[3] = ((unsigned)(ga >> 32) & 0x01FFFFFFu)      // global_addr[56:32]
          | (2u << 30);                             // type = 2 ("image")
  i32x8 g1;
  g1[0] = (1 << 16)        // data_size = 1 -> 2 bytes
        | (1 << 20)        // pad_enable
        | (7 << 22)        // pad_interval: every 256 dwords (one 1024B row)
        | (3 << 25);       // pad_amount: 4 dwords (16 B)
  g1[1] = (512 << 16);     // tensor_dim0 = 512 (bits 79:48, low half)
  g1[2] = (16 << 16);      // tensor_dim0 hi = 0 | tensor_dim1 = 16 (bits 111:80)
  g1[3] = (512 << 16);     // tensor_dim1 hi = 0 | tile_dim0 = 512 (bits 127:112)
  g1[4] = 16;              // tile_dim1 = 16, tile_dim2 = 0
  g1[5] = 512;             // tensor_dim0_stride (48b) low
  g1[6] = 0;               // stride hi | tensor_dim1_stride low
  g1[7] = 0;
  i32x4 z4 = {0, 0, 0, 0};
  i32x8 z8 = {0, 0, 0, 0, 0, 0, 0, 0};
  __builtin_amdgcn_tensor_load_to_lds(g0, g1, z4, z4, z8, 0);
}

// ---------------- kernel 1a: embedding gather -> padded bf16 [B][TPAD][E] ----
__global__ __launch_bounds__(256) void k_gather(const int* __restrict__ ids,
                                                const float* __restrict__ embed_w,
                                                unsigned short* __restrict__ xpad) {
  const long n = (long)BB * TPAD * E;
  for (long i = (long)blockIdx.x * blockDim.x + threadIdx.x; i < n;
       i += (long)gridDim.x * blockDim.x) {
    int e = (int)(i % E);
    int t = (int)((i / E) % TPAD);
    int b = (int)(i / ((long)E * TPAD));
    int ts = t - PAD;
    float v = 0.f;
    if (ts >= 0 && ts < TT) {
      int id = ids[(long)b * TT + ts];
      v = embed_w[(long)id * E + e];
    }
    xpad[i] = f2bf(v);
  }
}

// ---------------- kernel 1b: pack conv_w [K,E,KS] -> Wc bf16 [K][KS*E] -------
__global__ __launch_bounds__(256) void k_packw(const float* __restrict__ conv_w,
                                               unsigned short* __restrict__ Wc) {
  const long n = (long)KK * JTOT;
  for (long i = (long)blockIdx.x * blockDim.x + threadIdx.x; i < n;
       i += (long)gridDim.x * blockDim.x) {
    int e  = (int)(i % E);
    int jk = (int)((i / E) % KS);
    int k  = (int)(i / JTOT);
    Wc[i] = f2bf(conv_w[((long)k * E + e) * KS + jk]);
  }
}

// ---------------- kernel 1c: generic f32 -> bf16 convert ---------------------
__global__ __launch_bounds__(256) void k_cvt(const float* __restrict__ src,
                                             unsigned short* __restrict__ dst,
                                             long n) {
  for (long i = (long)blockIdx.x * blockDim.x + threadIdx.x; i < n;
       i += (long)gridDim.x * blockDim.x)
    dst[i] = f2bf(src[i]);
}

// ---------------- kernel 2: conv as WMMA GEMM + bias + ReLU ------------------
__global__ __launch_bounds__(256) void k_conv(const unsigned short* __restrict__ xpad,
                                              const unsigned short* __restrict__ Wc,
                                              const float* __restrict__ conv_b,
                                              unsigned short* __restrict__ feats) {
  const int wave = blockIdx.x * (blockDim.x >> 5) + (threadIdx.x >> 5);
  const int lane = threadIdx.x & 31;
  const int row  = lane & 15;
  const int hi   = lane >> 4;
  const int k0 = (wave & 31) << 4;            // 32 k-tiles
  const int t0 = ((wave >> 5) & 127) << 4;    // 128 t-tiles
  const int b  = wave >> 12;                  // 8 batches

  const unsigned short* arow = xpad + ((size_t)b * TPAD + t0 + row) * E;
  const unsigned short* bcol = Wc + (size_t)(k0 + row) * JTOT;

  v8f c = {};
#pragma unroll 8
  for (int jc = 0; jc < JCH; ++jc) {
    v16bf a = load_frag(arow, jc * 32, hi);
    v16bf w = load_frag(bcol, jc * 32, hi);
    c = __builtin_amdgcn_wmma_f32_16x16x32_bf16(false, a, false, w, (short)0, c,
                                                false, false);
  }
  const int k = k0 + row;
  const float bias = conv_b[k];
#pragma unroll
  for (int r = 0; r < 8; ++r) {
    int t = t0 + hi * 8 + r;
    float v = c[r] + bias;
    v = v > 0.f ? v : 0.f;
    feats[((size_t)b * TT + t) * KK + k] = f2bf(v);
  }
}

// ---------------- kernel 3: fused dual-GEMM + online softmax -----------------
// Block = 8 waves, one batch, 8 label-tiles. feats tiles DMA'd to LDS by the
// Tensor Data Mover (double-buffered, TENSORcnt), shared by all 8 waves.
__global__ __launch_bounds__(256) void k_attn(const unsigned short* __restrict__ Ubf,
                                              const unsigned short* __restrict__ Fbf,
                                              const unsigned short* __restrict__ feats,
                                              const float* __restrict__ final_b,
                                              float* __restrict__ yhat) {
  __shared__ unsigned short sbuf[2][LTILE];

  const int wv   = threadIdx.x >> 5;
  const int lane = threadIdx.x & 31;
  const int row  = lane & 15;
  const int hi   = lane >> 4;
  const int b  = blockIdx.x >> 6;                       // 64 blocks per batch
  const int l0 = (((blockIdx.x & 63) << 3) + wv) << 4;  // this wave's 16 labels

  // Register-resident A fragments: U and final_w tiles, read from HBM once.
  v16bf uf[KCH], ff[KCH];
  const unsigned short* urow = Ubf + (size_t)(l0 + row) * KK;
  const unsigned short* frow = Fbf + (size_t)(l0 + row) * KK;
#pragma unroll
  for (int kc = 0; kc < KCH; ++kc) {
    uf[kc] = load_frag(urow, kc * 32, hi);
    ff[kc] = load_frag(frow, kc * 32, hi);
  }

  float m[8], num[8], den[8];
#pragma unroll
  for (int r = 0; r < 8; ++r) { m[r] = -3.0e38f; num[r] = 0.f; den[r] = 0.f; }

  const unsigned short* fbase = feats + (size_t)b * TT * KK;
  const unsigned lds0 = (unsigned)(uintptr_t)&sbuf[0][0];
  const unsigned lds1 = (unsigned)(uintptr_t)&sbuf[1][0];

  // Prologue: DMA tile 0 into buffer 0.
  if (wv == 0) tdm_load_tile(fbase, lds0);

  const int NT = TT / 16;                               // 128 t-tiles
  for (int ti = 0; ti < NT; ++ti) {
    if (wv == 0) {
      if (ti + 1 < NT) {
        // Prefetch next tile into the other buffer, then wait for tile ti.
        tdm_load_tile(fbase + (size_t)(ti + 1) * 16 * KK,
                      (ti & 1) ? lds0 : lds1);
        __builtin_amdgcn_s_wait_tensorcnt(1);
      } else {
        __builtin_amdgcn_s_wait_tensorcnt(0);
      }
    }
    __syncthreads();                                    // tile ti visible to all

    const unsigned short* bcol = &sbuf[ti & 1][row * LROW];
    v8f sc = {};
    v8f vc = {};
#pragma unroll
    for (int kc = 0; kc < KCH; ++kc) {
      v16bf bf = load_frag(bcol, kc * 32, hi);          // ds_load_b128 x2
      sc = __builtin_amdgcn_wmma_f32_16x16x32_bf16(false, uf[kc], false, bf,
                                                   (short)0, sc, false, false);
      vc = __builtin_amdgcn_wmma_f32_16x16x32_bf16(false, ff[kc], false, bf,
                                                   (short)0, vc, false, false);
    }
    // online softmax: each lane owns column (16*ti + row) for its 8 rows
#pragma unroll
    for (int r = 0; r < 8; ++r) {
      float s = sc[r], v = vc[r];
      float nm = fmaxf(m[r], s);
      float s0 = __expf(m[r] - nm);
      float es = __expf(s - nm);
      num[r] = num[r] * s0 + es * v;
      den[r] = den[r] * s0 + es;
      m[r] = nm;
    }
    __syncthreads();                 // everyone done with buf[ti&1] before reuse
  }

  // cross-lane merge within each 16-lane column group (wave32)
#pragma unroll
  for (int off = 1; off < 16; off <<= 1) {
#pragma unroll
    for (int r = 0; r < 8; ++r) {
      float om = __shfl_xor(m[r], off, 32);
      float on = __shfl_xor(num[r], off, 32);
      float od = __shfl_xor(den[r], off, 32);
      float nm = fmaxf(m[r], om);
      float s0 = __expf(m[r] - nm);
      float s1 = __expf(om - nm);
      num[r] = num[r] * s0 + on * s1;
      den[r] = den[r] * s0 + od * s1;
      m[r] = nm;
    }
  }

  if (row == 0) {
#pragma unroll
    for (int r = 0; r < 8; ++r) {
      int l = l0 + hi * 8 + r;
      yhat[(size_t)b * LL + l] = num[r] / den[r] + final_b[l];
    }
  }
}

// ---------------- kernel 4: BCE-with-logits mean -----------------------------
__global__ void k_loss_init(float* loss) { *loss = 0.f; }

__global__ __launch_bounds__(256) void k_loss(const float* __restrict__ yhat,
                                              const float* __restrict__ target,
                                              float* __restrict__ loss) {
  __shared__ float sred[256];
  const long n = (long)BB * LL;
  float acc = 0.f;
  for (long i = (long)blockIdx.x * blockDim.x + threadIdx.x; i < n;
       i += (long)gridDim.x * blockDim.x) {
    float y = yhat[i], t = target[i];
    acc += fmaxf(y, 0.f) - y * t + log1pf(__expf(-fabsf(y)));
  }
  sred[threadIdx.x] = acc;
  __syncthreads();
  for (int s = 128; s > 0; s >>= 1) {
    if (threadIdx.x < s) sred[threadIdx.x] += sred[threadIdx.x + s];
    __syncthreads();
  }
  if (threadIdx.x == 0) atomicAdd(loss, sred[0] / (float)n);
}

// ---------------- launcher ---------------------------------------------------
extern "C" void kernel_launch(void* const* d_in, const int* in_sizes, int n_in,
                              void* d_out, int out_size, void* d_ws, size_t ws_size,
                              hipStream_t stream) {
  (void)in_sizes; (void)n_in; (void)out_size; (void)ws_size;
  const int*   ids     = (const int*)  d_in[0];
  // d_in[1] = mask (unused), d_in[2] = G (unused)
  const float* target  = (const float*)d_in[3];
  const float* embed_w = (const float*)d_in[4];
  const float* conv_w  = (const float*)d_in[5];
  const float* conv_b  = (const float*)d_in[6];
  const float* U_w     = (const float*)d_in[7];
  const float* final_w = (const float*)d_in[8];
  const float* final_b = (const float*)d_in[9];

  char* ws = (char*)d_ws;
  size_t off = 0;
  unsigned short* xpad  = (unsigned short*)(ws + off); off += (size_t)BB * TPAD * E * 2;
  unsigned short* Wc    = (unsigned short*)(ws + off); off += (size_t)KK * JTOT * 2;
  unsigned short* Ubf   = (unsigned short*)(ws + off); off += (size_t)LL * KK * 2;
  unsigned short* Fbf   = (unsigned short*)(ws + off); off += (size_t)LL * KK * 2;
  unsigned short* feats = (unsigned short*)(ws + off); off += (size_t)BB * TT * KK * 2;

  float* yhat = (float*)d_out;
  float* loss = yhat + (size_t)BB * LL;

  {
    long n = (long)BB * TPAD * E;
    k_gather<<<dim3((unsigned)((n + 255) / 256)), dim3(256), 0, stream>>>(ids, embed_w, xpad);
  }
  {
    long n = (long)KK * JTOT;
    k_packw<<<dim3((unsigned)((n + 255) / 256)), dim3(256), 0, stream>>>(conv_w, Wc);
  }
  {
    long n = (long)LL * KK;
    k_cvt<<<dim3((unsigned)((n + 255) / 256)), dim3(256), 0, stream>>>(U_w, Ubf, n);
    k_cvt<<<dim3((unsigned)((n + 255) / 256)), dim3(256), 0, stream>>>(final_w, Fbf, n);
  }
  // conv: B * (T/16) * (K/16) = 32768 tiles, 8 waves/block -> 4096 blocks
  k_conv<<<dim3(4096), dim3(256), 0, stream>>>(xpad, Wc, conv_b, feats);
  // attention: B * (L/16) = 4096 waves, 8 waves/block -> 512 blocks
  k_attn<<<dim3(512), dim3(256), 0, stream>>>(Ubf, Fbf, feats, final_b, yhat);

  k_loss_init<<<dim3(1), dim3(1), 0, stream>>>(loss);
  k_loss<<<dim3(64), dim3(256), 0, stream>>>(yhat, target, loss);
}
